// FeatureNet_38345468019209
// MI455X (gfx1250) — matrix operand order repaired and checked by reading, not compiled, not run on hardware
//
#include <hip/hip_runtime.h>

// Problem constants (fixed by the reference's setup_inputs)
#define B_    16
#define N_    2048
#define DIM   128
#define KNN   8
#define M_    (B_ * KNN * N_)   // 262144 rows
#define EPSBN 1e-5f

#define AS1 __attribute__((address_space(1)))
#define AS3 __attribute__((address_space(3)))

typedef __attribute__((ext_vector_type(16))) _Float16 v16h;
typedef __attribute__((ext_vector_type(8)))  _Float16 v8h;
typedef __attribute__((ext_vector_type(8)))  float    v8f;
typedef int v4i_ __attribute__((vector_size(16)));   // matches builtin param

// ---------------------------------------------------------------------------
// 1) kNN: per batch, stage all 2048 points (3 channels) in LDS, each thread
//    maintains its top-8 (smallest squared distance, tie -> lower index).
// ---------------------------------------------------------------------------
__global__ __launch_bounds__(256) void knn_kernel(const float* __restrict__ x,
                                                  int* __restrict__ idx) {
    __shared__ float sx[3][N_];                 // 24 KB
    const int b     = blockIdx.x >> 3;          // 8 chunks per batch
    const int chunk = blockIdx.x & 7;
    const float* xb = x + (size_t)b * 3 * N_;
    for (int i = threadIdx.x; i < N_; i += 256) {
        sx[0][i] = xb[i];
        sx[1][i] = xb[N_ + i];
        sx[2][i] = xb[2 * N_ + i];
    }
    __syncthreads();

    const int n = chunk * 256 + threadIdx.x;
    const float px = sx[0][n], py = sx[1][n], pz = sx[2][n];

    float bd[KNN];
    int   bi[KNN];
#pragma unroll
    for (int t = 0; t < KNN; ++t) { bd[t] = 3.0e38f; bi[t] = 0x7fffffff; }

    for (int m = 0; m < N_; ++m) {
        const float dx = sx[0][m] - px;
        const float dy = sx[1][m] - py;
        const float dz = sx[2][m] - pz;
        const float dist = dx * dx + dy * dy + dz * dz;
        if (dist < bd[KNN - 1] || (dist == bd[KNN - 1] && m < bi[KNN - 1])) {
            int t = KNN - 1;
            while (t > 0 && (dist < bd[t - 1] ||
                             (dist == bd[t - 1] && m < bi[t - 1]))) {
                bd[t] = bd[t - 1]; bi[t] = bi[t - 1]; --t;
            }
            bd[t] = dist; bi[t] = m;
        }
    }
    int* o = idx + ((size_t)b * N_ + n) * KNN;
#pragma unroll
    for (int t = 0; t < KNN; ++t) o[t] = bi[t];
}

// ---------------------------------------------------------------------------
// 2) edge features + layer1 (K=3 -> VALU, not worth WMMA), store h1 as fp16
//    with b128 vector stores. Row m = ((b*8 + kk)*N + n); layout [M, 128].
// ---------------------------------------------------------------------------
__global__ __launch_bounds__(256) void edge_h1_kernel(
    const float* __restrict__ x, const int* __restrict__ idx,
    const float* __restrict__ W1, const float* __restrict__ b1,
    _Float16* __restrict__ h1) {
    __shared__ float sW[DIM * 3];
    __shared__ float sb[DIM];
    for (int i = threadIdx.x; i < DIM * 3; i += 256) sW[i] = W1[i];
    if (threadIdx.x < DIM) sb[threadIdx.x] = b1[threadIdx.x];
    __syncthreads();

    const long gid = (long)blockIdx.x * 256 + threadIdx.x;   // < M_
    const int n  = (int)(gid % N_);
    const long t = gid / N_;
    const int kk = (int)(t % KNN);
    const int b  = (int)(t / KNN);

    const float* xb = x + (size_t)b * 3 * N_;
    const int j = idx[((size_t)b * N_ + n) * KNN + kk];
    const float e0 = xb[j]          - xb[n];
    const float e1 = xb[N_ + j]     - xb[N_ + n];
    const float e2 = xb[2 * N_ + j] - xb[2 * N_ + n];

    _Float16* outp = h1 + (size_t)gid * DIM;
#pragma unroll 4
    for (int dv = 0; dv < DIM / 8; ++dv) {
        v8h pack;
#pragma unroll
        for (int jj = 0; jj < 8; ++jj) {
            const int d = dv * 8 + jj;
            pack[jj] = (_Float16)(sW[d * 3] * e0 + sW[d * 3 + 1] * e1 +
                                  sW[d * 3 + 2] * e2 + sb[d]);
        }
        *(v8h*)(outp + dv * 8) = pack;
    }
}

// ---------------------------------------------------------------------------
// 3) per-channel sum / sumsq for BatchNorm (fp32 accumulation, f32 atomics)
// ---------------------------------------------------------------------------
__global__ void zero_kernel(float* __restrict__ p) { p[threadIdx.x] = 0.0f; }

__global__ __launch_bounds__(128) void stats_kernel(
    const _Float16* __restrict__ h, float* __restrict__ sums) {
    const int d = threadIdx.x;              // channel
    float s = 0.f, sq = 0.f;
    for (long m = blockIdx.x; m < M_; m += gridDim.x) {
        const float v = (float)h[m * DIM + d];
        s += v; sq += v * v;
    }
    atomicAdd(&sums[d], s);
    atomicAdd(&sums[DIM + d], sq);
}

__global__ void bn_params_kernel(const float* __restrict__ sums,
                                 const float* __restrict__ g,
                                 const float* __restrict__ be,
                                 float* __restrict__ prm) {
    const int d = threadIdx.x;
    const float invM = 1.0f / (float)M_;
    const float mean = sums[d] * invM;
    const float var  = sums[DIM + d] * invM - mean * mean;
    const float a = g[d] * rsqrtf(var + EPSBN);
    prm[d]       = a;
    prm[DIM + d] = be[d] - mean * a;
}

__global__ void f32_to_f16_kernel(const float* __restrict__ src,
                                  _Float16* __restrict__ dst, int n) {
    const int i = blockIdx.x * 256 + threadIdx.x;
    if (i < n) dst[i] = (_Float16)src[i];
}

// ---------------------------------------------------------------------------
// 4) Fused BN+ReLU(prev) -> GEMM (128x128, f16 in / f32 acc via WMMA) -> +bias
//    Block = 256 threads = 8 waves; wave owns 16 rows x all 128 channels
//    (8 accumulator tiles, 4 K-chunks of 32 -> 32 v_wmma per wave).
//    W preload uses the CDNA5 async global->LDS path (ASYNCcnt) when the
//    toolchain exposes it. Epilogue stages the D-tiles through LDS (reusing
//    the weight buffer) so the output stream is coalesced b128 stores.
// ---------------------------------------------------------------------------
__global__ __launch_bounds__(256) void gemm_bn_kernel(
    const _Float16* __restrict__ hin,   // [M,128] pre-BN
    const float* __restrict__ prm,      // a[128], c[128]
    const _Float16* __restrict__ Wh,    // [128,128] row-major [d][c]
    const float* __restrict__ bias,     // [128]
    _Float16* __restrict__ hout) {      // [M,128] pre-BN of next layer
    __shared__ _Float16 sW[DIM * DIM];  // 32 KB (weights, then D-tile staging)
    __shared__ float sA[DIM], sC[DIM];

    const int tid = threadIdx.x;
    // ---- weights -> LDS ----
#if defined(__gfx1250__) && __has_builtin(__builtin_amdgcn_global_load_async_to_lds_b128)
    {
        AS1 v4i_* g = (AS1 v4i_*)(const void*)Wh;   // 16B elements, global AS
        AS3 v4i_* l = (AS3 v4i_*)(void*)sW;         // 16B elements, LDS AS
#pragma unroll
        for (int i = 0; i < 8; ++i)                 // 256 thr * 8 * 16B = 32 KB
            __builtin_amdgcn_global_load_async_to_lds_b128(
                g + (tid + i * 256), l + (tid + i * 256), 0, 0);
#if __has_builtin(__builtin_amdgcn_s_wait_asynccnt)
        __builtin_amdgcn_s_wait_asynccnt(0);
#else
        asm volatile("s_wait_asynccnt 0x0" ::: "memory");
#endif
    }
#else
    {
        const uint4* gw = (const uint4*)Wh;
        uint4* sw = (uint4*)sW;
        for (int i = tid; i < (DIM * DIM) / 8; i += 256) sw[i] = gw[i];
    }
#endif
    if (tid < DIM) { sA[tid] = prm[tid]; sC[tid] = prm[DIM + tid]; }
    __syncthreads();

    const int wave   = tid >> 5;
    const int lane   = tid & 31;
    const int laneLo = lane & 15;
    const int hi     = lane >> 4;                     // 0|1
    const size_t mtile = (size_t)blockIdx.x * 128 + (size_t)wave * 16;
    const _Float16* arow = hin + (mtile + laneLo) * DIM;

    v8f acc[8];
#pragma unroll
    for (int t = 0; t < 8; ++t)
#pragma unroll
        for (int r = 0; r < 8; ++r) acc[t][r] = 0.0f;

    // streaming hint for the next block's activation stripe
    __builtin_prefetch(arow + (size_t)128 * DIM, 0, 1);

#pragma unroll
    for (int kc = 0; kc < 4; ++kc) {
        // A operand: lane<16 -> K {kc*32+0..7, +16..23}; lane>=16 -> +8 shift
        const int c0 = kc * 32 + hi * 8;
        const v8h a0 = *(const v8h*)(arow + c0);
        const v8h a1 = *(const v8h*)(arow + c0 + 16);
        v16h Aop;
#pragma unroll
        for (int j = 0; j < 8; ++j) {
            const float v0 = (float)a0[j] * sA[c0 + j]      + sC[c0 + j];
            const float v1 = (float)a1[j] * sA[c0 + 16 + j] + sC[c0 + 16 + j];
            Aop[j]     = (_Float16)fmaxf(v0, 0.0f);
            Aop[j + 8] = (_Float16)fmaxf(v1, 0.0f);
        }
#pragma unroll
        for (int dt = 0; dt < 8; ++dt) {
            // B operand: lane = column d, K {kc*32+hi*8+0..7, +16..23} from W row d
            const int d  = dt * 16 + laneLo;
            const int cb = kc * 32 + hi * 8;
            const _Float16* wrow = &sW[d * DIM + cb];
            const v8h b0 = *(const v8h*)(wrow);
            const v8h b1 = *(const v8h*)(wrow + 16);
            const v16h Bop = __builtin_shufflevector(
                b0, b1, 0, 1, 2, 3, 4, 5, 6, 7, 8, 9, 10, 11, 12, 13, 14, 15);
            acc[dt] = __builtin_amdgcn_wmma_f32_16x16x32_f16(
                false, Aop, false, Bop, (short)0, acc[dt], false, false);
        }
    }

    // ---- epilogue: stage tiles in LDS (W is dead), coalesced b128 stores ----
    __syncthreads();                       // everyone done reading weights
    _Float16* stile = sW + wave * (16 * DIM);   // 4 KB per wave
#pragma unroll
    for (int dt = 0; dt < 8; ++dt) {
        const int d = dt * 16 + laneLo;
        const float bv = bias[d];
#pragma unroll
        for (int r = 0; r < 8; ++r)        // lane holds col d, rows hi*8+r
            stile[(hi * 8 + r) * DIM + d] = (_Float16)(acc[dt][r] + bv);
    }
    // same-wave LDS ops are in order: read back rows, write coalesced
#pragma unroll
    for (int it = 0; it < 8; ++it) {
        const int row = it * 2 + hi;
        const v8h val = *(const v8h*)&stile[row * DIM + laneLo * 8];
        *(v8h*)&hout[(mtile + row) * DIM + laneLo * 8] = val;
    }
}

// ---------------------------------------------------------------------------
// 5) BN3 + ReLU + max over k=8 -> out [B,128,N] fp32.
//    Block = one (batch, 64-point tile): coalesced f16 row reads, LDS
//    transpose (padded), coalesced float2 stores along N.
// ---------------------------------------------------------------------------
__global__ __launch_bounds__(256) void final_kernel(
    const _Float16* __restrict__ h3, const float* __restrict__ prm,
    float* __restrict__ out) {
    __shared__ float sbuf[64][DIM + 1];    // ~33 KB, padded stride
    const int b  = blockIdx.x >> 5;        // 32 tiles of 64 points per batch
    const int n0 = (blockIdx.x & 31) * 64;
    const int d  = threadIdx.x & 127;
    const int nl = threadIdx.x >> 7;       // 0|1
    const float a = prm[d], c = prm[DIM + d];

    for (int np = 0; np < 64; np += 2) {
        const int n = n0 + np + nl;
        float best = 0.0f;                 // ReLU => >= 0
#pragma unroll
        for (int kk = 0; kk < KNN; ++kk) {
            const size_t m = ((size_t)(b * KNN + kk) * N_ + n);
            best = fmaxf(best, fmaxf((float)h3[m * DIM + d] * a + c, 0.0f));
        }
        sbuf[np + nl][d] = best;
    }
    __syncthreads();

    const int wv = threadIdx.x >> 5, ln = threadIdx.x & 31;
#pragma unroll
    for (int i = 0; i < 16; ++i) {         // wave w covers d = w, w+8, ...
        const int dd = wv + i * 8;
        float2 v;
        v.x = sbuf[ln * 2 + 0][dd];
        v.y = sbuf[ln * 2 + 1][dd];
        *(float2*)&out[((size_t)b * DIM + dd) * N_ + n0 + ln * 2] = v;
    }
}

// ---------------------------------------------------------------------------
extern "C" void kernel_launch(void* const* d_in, const int* in_sizes, int n_in,
                              void* d_out, int out_size, void* d_ws,
                              size_t ws_size, hipStream_t stream) {
    const float* x   = (const float*)d_in[0];
    const float* W1  = (const float*)d_in[1];
    const float* b1  = (const float*)d_in[2];
    const float* g1  = (const float*)d_in[3];
    const float* be1 = (const float*)d_in[4];
    const float* W2  = (const float*)d_in[5];
    const float* b2  = (const float*)d_in[6];
    const float* g2  = (const float*)d_in[7];
    const float* be2 = (const float*)d_in[8];
    const float* W3  = (const float*)d_in[9];
    const float* b3  = (const float*)d_in[10];
    const float* g3  = (const float*)d_in[11];
    const float* be3 = (const float*)d_in[12];
    float* out = (float*)d_out;

    char* ws = (char*)d_ws;
    const size_t HBYTES = (size_t)M_ * DIM * sizeof(_Float16);  // 64 MB
    _Float16* hA   = (_Float16*)(ws);
    _Float16* hB   = (_Float16*)(ws + HBYTES);
    int*      idx  = (int*)(ws + 2 * HBYTES);
    _Float16* W2h  = (_Float16*)(ws + 2 * HBYTES + 4 * 1024 * 1024);
    _Float16* W3h  = (_Float16*)(ws + 2 * HBYTES + 4 * 1024 * 1024 + 65536);
    float*    sums = (float*)(ws + 2 * HBYTES + 4 * 1024 * 1024 + 131072);
    float*    prm  = (float*)(ws + 2 * HBYTES + 4 * 1024 * 1024 + 131072 + 4096);

    knn_kernel<<<B_ * 8, 256, 0, stream>>>(x, idx);
    f32_to_f16_kernel<<<(DIM * DIM) / 256, 256, 0, stream>>>(W2, W2h, DIM * DIM);
    f32_to_f16_kernel<<<(DIM * DIM) / 256, 256, 0, stream>>>(W3, W3h, DIM * DIM);

    // layer 1 (3 -> 128) + BN1 stats
    edge_h1_kernel<<<M_ / 256, 256, 0, stream>>>(x, idx, W1, b1, hA);
    zero_kernel<<<1, 256, 0, stream>>>(sums);
    stats_kernel<<<512, 128, 0, stream>>>(hA, sums);
    bn_params_kernel<<<1, 128, 0, stream>>>(sums, g1, be1, prm);

    // layer 2: BN1+ReLU fused into WMMA GEMM, then BN2 stats
    gemm_bn_kernel<<<M_ / 128, 256, 0, stream>>>(hA, prm, W2h, b2, hB);
    zero_kernel<<<1, 256, 0, stream>>>(sums);
    stats_kernel<<<512, 128, 0, stream>>>(hB, sums);
    bn_params_kernel<<<1, 128, 0, stream>>>(sums, g2, be2, prm);

    // layer 3
    gemm_bn_kernel<<<M_ / 128, 256, 0, stream>>>(hB, prm, W3h, b3, hA);
    zero_kernel<<<1, 256, 0, stream>>>(sums);
    stats_kernel<<<512, 128, 0, stream>>>(hA, sums);
    bn_params_kernel<<<1, 128, 0, stream>>>(sums, g3, be3, prm);

    // BN3 + ReLU + max over k
    final_kernel<<<B_ * 32, 256, 0, stream>>>(hA, prm, out);
}
